// Attention_62672162783289
// MI455X (gfx1250) — matrix operand order repaired and checked by reading, not compiled, not run on hardware
//
#include <hip/hip_runtime.h>
#include <hip/hip_bf16.h>
#include <math.h>

typedef __attribute__((ext_vector_type(16))) _Float16 v16h;
typedef __attribute__((ext_vector_type(8)))  _Float16 v8h;
typedef __attribute__((ext_vector_type(8)))  float    v8f;
typedef int v4i __attribute__((vector_size(16)));

#define B_    16
#define N_    1024
#define DIM_  768
#define H_    12
#define KD_   64
#define QKV3_ 2304
#define SCALE_ 0.125f
#define LN_EPS_ 1e-5f

#if defined(__gfx1250__) && __has_builtin(__builtin_amdgcn_global_load_async_to_lds_b128)
#define HAVE_ASYNC 1
#else
#define HAVE_ASYNC 0
#endif

typedef __attribute__((address_space(1))) v4i* gptr_b128;
typedef __attribute__((address_space(3))) v4i* lptr_b128;

// ---------- WMMA helpers ----------------------------------------------------
static __device__ __forceinline__ v8f wmma_f16(v16h a, v16h b, v8f c) {
  // D = A(16x32 f16) * B(32x16 f16) + C(16x16 f32)
  return __builtin_amdgcn_wmma_f32_16x16x32_f16(false, a, false, b, (short)0, c,
                                                false, false);
}

// A-matrix 16x32 f16 fragment (ISA 7.12.2): lane r=lane&15 is row M=r,
// half-group g=lane>>4: f[0..7] = K=8g..8g+7, f[8..15] = K=16+8g..16+8g+7.
static __device__ __forceinline__ v16h load_frag_a(const _Float16* base, int ld) {
  int lane = threadIdx.x & 31;
  int r = lane & 15;
  int g = lane >> 4;
  const _Float16* p = base + (long)r * ld + g * 8;
  v8h a0 = *(const v8h*)(p);
  v8h a1 = *(const v8h*)(p + 16);
  v16h f;
#pragma unroll
  for (int i = 0; i < 8; i++) { f[i] = a0[i]; f[8 + i] = a1[i]; }
  return f;
}

// B-matrix (supplied as B^T, row-major N x K) fragment: lane r = column N,
// group g=lane>>4: f[0..7] = K=8g..8g+7, f[8..15] = K=16+8g..16+8g+7.
static __device__ __forceinline__ v16h load_frag_b(const _Float16* base, int ld) {
  int lane = threadIdx.x & 31;
  int r = lane & 15;
  int g = lane >> 4;
  const _Float16* p = base + (long)r * ld + g * 8;
  v8h a0 = *(const v8h*)(p);
  v8h a1 = *(const v8h*)(p + 16);
  v16h f;
#pragma unroll
  for (int i = 0; i < 8; i++) { f[i] = a0[i]; f[8 + i] = a1[i]; }
  return f;
}

static __device__ __forceinline__ v8f zero8() {
  v8f z = {0.f, 0.f, 0.f, 0.f, 0.f, 0.f, 0.f, 0.f};
  return z;
}

// one 16-byte global -> LDS chunk, async if available
static __device__ __forceinline__ void copy16(const _Float16* src, _Float16* dst) {
#if HAVE_ASYNC
  __builtin_amdgcn_global_load_async_to_lds_b128(
      (gptr_b128)(void*)const_cast<_Float16*>(src), (lptr_b128)(void*)dst, 0, 0);
#else
  *(v8h*)dst = *(const v8h*)src;
#endif
}

static __device__ __forceinline__ void wait_async() {
#if HAVE_ASYNC
  asm volatile("s_wait_asynccnt 0x0" ::: "memory");
#endif
}

// ---------- prep: f16 weights + bias table ----------------------------------
__global__ void prep_kernel(const float* __restrict__ qkv_w,
                            const float* __restrict__ proj_w,
                            const float* __restrict__ attn_biases,
                            const int*   __restrict__ bias_idxs,
                            _Float16* wq, _Float16* wp, _Float16* bm) {
  const long nwq = (long)QKV3_ * DIM_;
  const long nwp = (long)DIM_ * DIM_;
  const long nbm = (long)H_ * N_ * N_;
  const long total = nwq + nwp + nbm;
  for (long i = (long)blockIdx.x * blockDim.x + threadIdx.x; i < total;
       i += (long)gridDim.x * blockDim.x) {
    if (i < nwq) {
      wq[i] = (_Float16)qkv_w[i];
    } else if (i < nwq + nwp) {
      long j = i - nwq;
      wp[j] = (_Float16)proj_w[j];
    } else {
      long j = i - nwq - nwp;          // j = h*N*N + n*N + m
      long h = j >> 20;                // / (1024*1024)
      long nm = j & ((1L << 20) - 1);
      bm[j] = (_Float16)attn_biases[h * N_ + bias_idxs[nm]];
    }
  }
}

// ---------- LayerNorm -> f16 -------------------------------------------------
__global__ __launch_bounds__(256) void ln_kernel(const float* __restrict__ x,
                                                 const float* __restrict__ ln_w,
                                                 const float* __restrict__ ln_b,
                                                 _Float16* __restrict__ xh) {
  int row = blockIdx.x;                       // B*N rows
  const float* xr = x + (long)row * DIM_;
  int tid = threadIdx.x;
  float v0 = xr[tid], v1 = xr[tid + 256], v2 = xr[tid + 512];
  float s = v0 + v1 + v2;
  float s2 = v0 * v0 + v1 * v1 + v2 * v2;
#pragma unroll
  for (int m = 1; m < 32; m <<= 1) {
    s += __shfl_xor(s, m, 32);
    s2 += __shfl_xor(s2, m, 32);
  }
  __shared__ float ps[8], ps2[8];
  int w = tid >> 5, lane = tid & 31;
  if (lane == 0) { ps[w] = s; ps2[w] = s2; }
  __syncthreads();
  if (w == 0) {
    float a = (lane < 8) ? ps[lane] : 0.f;
    float b = (lane < 8) ? ps2[lane] : 0.f;
#pragma unroll
    for (int m = 1; m < 8; m <<= 1) {
      a += __shfl_xor(a, m, 32);
      b += __shfl_xor(b, m, 32);
    }
    if (lane == 0) { ps[0] = a; ps2[0] = b; }
  }
  __syncthreads();
  float mu = ps[0] * (1.0f / DIM_);
  float var = ps2[0] * (1.0f / DIM_) - mu * mu;
  float rstd = rsqrtf(var + LN_EPS_);
  _Float16* xo = xh + (long)row * DIM_;
#pragma unroll
  for (int i = tid; i < DIM_; i += 256) {
    float xn = (xr[i] - mu) * rstd * ln_w[i] + ln_b[i];
    xo[i] = (_Float16)xn;
  }
}

// ---------- shared double-buffered 128x64 GEMM mainloop ----------------------
// A: (rows x 768) f16 row-major, Bt: (cols x 768) f16 row-major.
// Block = 256 threads / 8 waves; wave (wm,wn) owns a 32x32 sub-tile.
#define ATILE_ (128 * 40)
#define BTILE_ (64 * 40)

static __device__ __forceinline__ void load_tiles(const _Float16* Ag,
                                                  const _Float16* Bg, int k0,
                                                  _Float16* Asb, _Float16* Bsb) {
  int tid = threadIdx.x;
#pragma unroll
  for (int s = 0; s < 2; s++) {           // A: 128 rows x 32 halves
    int id = tid + 256 * s;
    int r = id >> 2, c = (id & 3) * 8;
    copy16(Ag + (long)r * DIM_ + k0 + c, Asb + r * 40 + c);
  }
  {                                       // B: 64 rows x 32 halves
    int r = tid >> 2, c = (tid & 3) * 8;
    copy16(Bg + (long)r * DIM_ + k0 + c, Bsb + r * 40 + c);
  }
}

static __device__ __forceinline__ void gemm_loop(const _Float16* Ag,
                                                 const _Float16* Bg,
                                                 _Float16* As, _Float16* Bs,
                                                 int wm, int wn,
                                                 v8f& c00, v8f& c01,
                                                 v8f& c10, v8f& c11) {
  load_tiles(Ag, Bg, 0, As, Bs);
  int buf = 0;
#pragma unroll
  for (int k0 = 0; k0 < DIM_; k0 += 32) {
    wait_async();                         // drain our own ASYNCcnt, then publish
    __syncthreads();
    if (k0 + 32 < DIM_)
      load_tiles(Ag, Bg, k0 + 32, As + (buf ^ 1) * ATILE_, Bs + (buf ^ 1) * BTILE_);
    _Float16* Asb = As + buf * ATILE_;
    _Float16* Bsb = Bs + buf * BTILE_;
    v16h a0 = load_frag_a(Asb + (wm * 32) * 40, 40);
    v16h a1 = load_frag_a(Asb + (wm * 32 + 16) * 40, 40);
    v16h b0 = load_frag_b(Bsb + (wn * 32) * 40, 40);
    v16h b1 = load_frag_b(Bsb + (wn * 32 + 16) * 40, 40);
    c00 = wmma_f16(a0, b0, c00);
    c01 = wmma_f16(a0, b1, c01);
    c10 = wmma_f16(a1, b0, c10);
    c11 = wmma_f16(a1, b1, c11);
    buf ^= 1;
  }
}

// ---------- QKV GEMM: (16384x768) x (768x2304) with scatter epilogue --------
__global__ __launch_bounds__(256) void qkv_gemm(const _Float16* __restrict__ xh,
                                                const _Float16* __restrict__ wq,
                                                const float* __restrict__ qkv_b,
                                                _Float16* __restrict__ q,
                                                _Float16* __restrict__ k,
                                                _Float16* __restrict__ vT) {
  __shared__ __attribute__((aligned(16))) _Float16 As[2 * ATILE_];
  __shared__ __attribute__((aligned(16))) _Float16 Bs[2 * BTILE_];
  int tm = blockIdx.x;                         // 0..127 (rows/128)
  int tn = blockIdx.y;                         // 0..35  (cols/64)
  int tid = threadIdx.x;
  int w = tid >> 5, lane = tid & 31;
  int wm = w & 3, wn = w >> 2;
  v8f c00 = zero8(), c01 = zero8(), c10 = zero8(), c11 = zero8();
  gemm_loop(xh + (long)tm * 128 * DIM_, wq + (long)tn * 64 * DIM_, As, Bs, wm, wn,
            c00, c01, c10, c11);
  int cn = lane & 15, g = lane >> 4;
#pragma unroll
  for (int ar = 0; ar < 2; ar++) {
#pragma unroll
    for (int bc = 0; bc < 2; bc++) {
      v8f acc = (ar == 0) ? (bc == 0 ? c00 : c01) : (bc == 0 ? c10 : c11);
      int col = tn * 64 + wn * 32 + bc * 16 + cn;     // 0..2303
      int h = col / 192, rem = col - h * 192;
      float bias = qkv_b[col];
      long grow = (long)tm * 128 + wm * 32 + ar * 16 + 8 * g;
#pragma unroll
      for (int v = 0; v < 8; v++) {
        long r = grow + v;
        int bb = (int)(r >> 10);
        int tok = (int)(r & 1023);
        float val = acc[v] + bias;
        long bh = (long)bb * H_ + h;
        if (rem < 64)
          q[(bh * N_ + tok) * KD_ + rem] = (_Float16)(val * SCALE_);
        else if (rem < 128)
          k[(bh * N_ + tok) * KD_ + (rem - 64)] = (_Float16)val;
        else
          vT[(bh * KD_ + (rem - 128)) * N_ + tok] = (_Float16)val;
      }
    }
  }
}

// ---------- Flash attention: 64 query rows / block, 4 waves -----------------
__global__ __launch_bounds__(128) void attn_kernel(const _Float16* __restrict__ q,
                                                   const _Float16* __restrict__ k,
                                                   const _Float16* __restrict__ vT,
                                                   const _Float16* __restrict__ bm,
                                                   _Float16* __restrict__ ao) {
  __shared__ __attribute__((aligned(16))) _Float16 Ks[64 * 72];
  __shared__ __attribute__((aligned(16))) _Float16 Vs[64 * 72];
  __shared__ __attribute__((aligned(16))) _Float16 Ps[4][16 * 72];
  int rt = blockIdx.x, h = blockIdx.y, b = blockIdx.z;
  int tid = threadIdx.x, w = tid >> 5, lane = tid & 31;
  int cn = lane & 15, g = lane >> 4;
  long bh = (long)b * H_ + h;
  const _Float16* qp = q + (bh * N_ + (long)rt * 64 + w * 16) * KD_;  // pre-scaled
  const _Float16* kp = k + bh * N_ * KD_;
  const _Float16* vp = vT + bh * KD_ * N_;
  const _Float16* bp = bm + ((long)h * N_ + rt * 64 + w * 16) * N_;

  v16h qf0 = load_frag_a(qp, KD_);
  v16h qf1 = load_frag_a(qp + 32, KD_);
  v8f O0 = zero8(), O1 = zero8(), O2 = zero8(), O3 = zero8();
  float m8[8], l8[8];
#pragma unroll
  for (int v = 0; v < 8; v++) { m8[v] = -3.0e38f; l8[v] = 0.f; }

  int lr = tid >> 1, lc = (tid & 1) * 32;
  for (int j = 0; j < 16; j++) {                 // key blocks of 64
    __syncthreads();                             // prior reads of Ks/Vs done
#pragma unroll
    for (int c8 = 0; c8 < 32; c8 += 8) {
      copy16(kp + (long)(j * 64 + lr) * KD_ + lc + c8, Ks + lr * 72 + lc + c8);
      copy16(vp + (long)lr * N_ + j * 64 + lc + c8, Vs + lr * 72 + lc + c8);
    }
    wait_async();                                // our async writes landed
    __syncthreads();                             // publish K/V tiles

    // S = Qs K^T + bias  (bias preloaded into accumulator)
    v8f s[4];
#pragma unroll
    for (int t = 0; t < 4; t++) {
      v8f c;
      const _Float16* bpt = bp + j * 64 + t * 16 + cn;
#pragma unroll
      for (int v = 0; v < 8; v++) c[v] = (float)bpt[(long)(v + 8 * g) * N_];
      c = wmma_f16(qf0, load_frag_b(Ks + t * 16 * 72, 72), c);
      c = wmma_f16(qf1, load_frag_b(Ks + t * 16 * 72 + 32, 72), c);
      s[t] = c;
    }

    // online softmax (rows v+8g, 16-lane reductions)
    float alpha[8], rs[8];
#pragma unroll
    for (int v = 0; v < 8; v++) {
      float x = fmaxf(fmaxf(s[0][v], s[1][v]), fmaxf(s[2][v], s[3][v]));
#pragma unroll
      for (int msk = 1; msk < 16; msk <<= 1) x = fmaxf(x, __shfl_xor(x, msk, 32));
      float mn = fmaxf(m8[v], x);
      alpha[v] = __expf(m8[v] - mn);
      m8[v] = mn;
      rs[v] = 0.f;
    }
#pragma unroll
    for (int t = 0; t < 4; t++) {
#pragma unroll
      for (int v = 0; v < 8; v++) {
        float p = __expf(s[t][v] - m8[v]);
        rs[v] += p;
        Ps[w][(v + 8 * g) * 72 + t * 16 + cn] = (_Float16)p;
      }
    }
#pragma unroll
    for (int v = 0; v < 8; v++) {
      float x = rs[v];
#pragma unroll
      for (int msk = 1; msk < 16; msk <<= 1) x += __shfl_xor(x, msk, 32);
      l8[v] = l8[v] * alpha[v] + x;
      O0[v] *= alpha[v]; O1[v] *= alpha[v]; O2[v] *= alpha[v]; O3[v] *= alpha[v];
    }
    __syncthreads();                             // P tiles visible

    // O += P @ V   (B^T = vT tile, d-major in LDS)
    v16h p0 = load_frag_a(Ps[w], 72);
    v16h p1 = load_frag_a(Ps[w] + 32, 72);
    O0 = wmma_f16(p0, load_frag_b(Vs + 0 * 16 * 72, 72), O0);
    O0 = wmma_f16(p1, load_frag_b(Vs + 0 * 16 * 72 + 32, 72), O0);
    O1 = wmma_f16(p0, load_frag_b(Vs + 1 * 16 * 72, 72), O1);
    O1 = wmma_f16(p1, load_frag_b(Vs + 1 * 16 * 72 + 32, 72), O1);
    O2 = wmma_f16(p0, load_frag_b(Vs + 2 * 16 * 72, 72), O2);
    O2 = wmma_f16(p1, load_frag_b(Vs + 2 * 16 * 72 + 32, 72), O2);
    O3 = wmma_f16(p0, load_frag_b(Vs + 3 * 16 * 72, 72), O3);
    O3 = wmma_f16(p1, load_frag_b(Vs + 3 * 16 * 72 + 32, 72), O3);
  }

  _Float16* aop = ao + ((long)b * N_ + rt * 64 + w * 16) * DIM_ + h * KD_;
#pragma unroll
  for (int t = 0; t < 4; t++) {
    v8f O = (t == 0) ? O0 : (t == 1) ? O1 : (t == 2) ? O2 : O3;
#pragma unroll
    for (int v = 0; v < 8; v++)
      aop[(long)(v + 8 * g) * DIM_ + t * 16 + cn] = (_Float16)(O[v] / l8[v]);
  }
}

// ---------- Output projection: (16384x768) x (768x768) -> f32 ---------------
__global__ __launch_bounds__(256) void proj_gemm(const _Float16* __restrict__ ao,
                                                 const _Float16* __restrict__ wp,
                                                 const float* __restrict__ proj_b,
                                                 float* __restrict__ out) {
  __shared__ __attribute__((aligned(16))) _Float16 As[2 * ATILE_];
  __shared__ __attribute__((aligned(16))) _Float16 Bs[2 * BTILE_];
  int tm = blockIdx.x;                          // 0..127
  int tn = blockIdx.y;                          // 0..11
  int tid = threadIdx.x;
  int w = tid >> 5, lane = tid & 31;
  int wm = w & 3, wn = w >> 2;
  v8f c00 = zero8(), c01 = zero8(), c10 = zero8(), c11 = zero8();
  gemm_loop(ao + (long)tm * 128 * DIM_, wp + (long)tn * 64 * DIM_, As, Bs, wm, wn,
            c00, c01, c10, c11);
  int cn = lane & 15, g = lane >> 4;
#pragma unroll
  for (int ar = 0; ar < 2; ar++) {
#pragma unroll
    for (int bc = 0; bc < 2; bc++) {
      v8f acc = (ar == 0) ? (bc == 0 ? c00 : c01) : (bc == 0 ? c10 : c11);
      int col = tn * 64 + wn * 32 + bc * 16 + cn;
      float bias = proj_b[col];
      long grow = (long)tm * 128 + wm * 32 + ar * 16 + 8 * g;
#pragma unroll
      for (int v = 0; v < 8; v++)
        out[(grow + v) * DIM_ + col] = acc[v] + bias;
    }
  }
}

// ---------- host launcher ----------------------------------------------------
extern "C" void kernel_launch(void* const* d_in, const int* in_sizes, int n_in,
                              void* d_out, int out_size, void* d_ws, size_t ws_size,
                              hipStream_t stream) {
  (void)in_sizes; (void)n_in; (void)out_size; (void)ws_size;
  const float* x           = (const float*)d_in[0];
  const float* ln_w        = (const float*)d_in[1];
  const float* ln_b        = (const float*)d_in[2];
  const float* qkv_w       = (const float*)d_in[3];
  const float* qkv_b       = (const float*)d_in[4];
  const float* proj_w      = (const float*)d_in[5];
  const float* proj_b      = (const float*)d_in[6];
  const float* attn_biases = (const float*)d_in[7];
  const int*   bias_idxs   = (const int*)d_in[8];
  float* out = (float*)d_out;

  char* p = (char*)d_ws;
  _Float16* xh = (_Float16*)p; p += (size_t)B_ * N_ * DIM_ * 2;       // 25.2 MB
  _Float16* wq = (_Float16*)p; p += (size_t)QKV3_ * DIM_ * 2;         //  3.5 MB
  _Float16* wp = (_Float16*)p; p += (size_t)DIM_ * DIM_ * 2;          //  1.2 MB
  _Float16* q  = (_Float16*)p; p += (size_t)B_ * H_ * N_ * KD_ * 2;   // 25.2 MB
  _Float16* k  = (_Float16*)p; p += (size_t)B_ * H_ * N_ * KD_ * 2;   // 25.2 MB
  _Float16* vT = (_Float16*)p; p += (size_t)B_ * H_ * KD_ * N_ * 2;   // 25.2 MB
  _Float16* ao = (_Float16*)p; p += (size_t)B_ * N_ * DIM_ * 2;       // 25.2 MB
  _Float16* bm = (_Float16*)p; p += (size_t)H_ * N_ * N_ * 2;         // 25.2 MB

  prep_kernel<<<2048, 256, 0, stream>>>(qkv_w, proj_w, attn_biases, bias_idxs,
                                        wq, wp, bm);
  ln_kernel<<<B_ * N_, 256, 0, stream>>>(x, ln_w, ln_b, xh);
  qkv_gemm<<<dim3(128, 36), 256, 0, stream>>>(xh, wq, qkv_b, q, k, vT);
  attn_kernel<<<dim3(16, H_, B_), 128, 0, stream>>>(q, k, vT, bm, ao);
  proj_gemm<<<dim3(128, 12), 256, 0, stream>>>(ao, wp, proj_b, out);
}
